// GAT_Net_84756884620004
// MI455X (gfx1250) — compile-verified, matrix operand order checked by reading, not run on hardware
//
#include <hip/hip_runtime.h>
#include <hip/hip_bf16.h>
#include <math.h>

typedef __attribute__((ext_vector_type(16))) _Float16 v16h;
typedef __attribute__((ext_vector_type(8)))  float    v8f;

// ---------------- wave32 reductions ----------------
__device__ __forceinline__ float wave_max32(float v) {
#pragma unroll
  for (int off = 16; off > 0; off >>= 1) v = fmaxf(v, __shfl_xor(v, off, 32));
  return v;
}
__device__ __forceinline__ float wave_sum32(float v) {
#pragma unroll
  for (int off = 16; off > 0; off >>= 1) v += __shfl_xor(v, off, 32);
  return v;
}

// ---------------- CSR build ----------------
__global__ void k_init_deg(int* deg, int n) {
  int i = blockIdx.x * blockDim.x + threadIdx.x;
  if (i < n) deg[i] = 1;  // self loop
}
__global__ void k_count(const long long* __restrict__ dst, int e, int* deg) {
  int i = blockIdx.x * blockDim.x + threadIdx.x;
  if (i < e) atomicAdd(&deg[(int)dst[i]], 1);
}
__global__ void k_scan_block(const int* __restrict__ deg, int* __restrict__ rp,
                             int* __restrict__ bsum, int n) {
  __shared__ int s[1024];
  int tid = threadIdx.x;
  int i = blockIdx.x * 1024 + tid;
  int v = (i < n) ? deg[i] : 0;
  s[tid] = v;
  __syncthreads();
#pragma unroll
  for (int off = 1; off < 1024; off <<= 1) {
    int t = (tid >= off) ? s[tid - off] : 0;
    __syncthreads();
    s[tid] += t;
    __syncthreads();
  }
  if (i < n) rp[i] = s[tid] - v;  // exclusive
  if (tid == 1023) bsum[blockIdx.x] = s[1023];
}
__global__ void k_scan_bsum(int* bsum, int nb) {
  if (blockIdx.x == 0 && threadIdx.x == 0) {
    int acc = 0;
    for (int i = 0; i < nb; ++i) { int t = bsum[i]; bsum[i] = acc; acc += t; }
  }
}
__global__ void k_scan_add(int* rp, const int* __restrict__ bsum, int n, int total) {
  int i = blockIdx.x * blockDim.x + threadIdx.x;
  if (i < n) rp[i] += bsum[i >> 10];
  if (i == 0) rp[n] = total;
}
__global__ void k_copy_i32(const int* __restrict__ a, int* __restrict__ b, int n) {
  int i = blockIdx.x * blockDim.x + threadIdx.x;
  if (i < n) b[i] = a[i];
}
__global__ void k_fill_csr(const long long* __restrict__ src, const long long* __restrict__ dst,
                           int e, int n, int* cur, int* __restrict__ ci) {
  int i = blockIdx.x * blockDim.x + threadIdx.x;
  if (i < e) {
    int d = (int)dst[i];
    int p = atomicAdd(&cur[d], 1);
    ci[p] = (int)src[i];
  } else if (i < e + n) {
    int v = i - e;
    int p = atomicAdd(&cur[v], 1);
    ci[p] = v;  // self loop
  }
}

// ---------------- WMMA GEMM: Out[M,NOUT] = A[M,K] * W[K,NOUT], f16 mul / f32 acc ----
// One wave owns a full 16-row M-tile and ALL N-tiles (A loaded once, reused NT times).
// W is staged per-block into LDS, pre-converted to f16 and pre-swizzled into the
// per-lane WMMA-B fragment layout, so the inner loop is ds_load_b128 + v_wmma.
template <int K, int NOUT>
__global__ void k_gemm_f16wmma(const float* __restrict__ A, const float* __restrict__ W,
                               float* __restrict__ Out, int M) {
  constexpr int NT = NOUT / 16;  // N tiles
  constexpr int KT = K / 32;     // K steps
  __shared__ __attribute__((aligned(32))) _Float16 w_lds[KT * NT * 512];

  // Cooperative stage: element t of fragment layout <- W[k][n], converted to f16.
  for (int t = threadIdx.x; t < KT * NT * 512; t += blockDim.x) {
    int e = t & 15;          // element within lane fragment
    int L = (t >> 4) & 31;   // lane
    int tile = t >> 9;       // fragment index
    int tn = tile % NT;
    int kbi = tile / NT;
    int v = e >> 1;
    int k = kbi * 32 + (L >> 4) * 8 + ((v < 4) ? 2 * v : 16 + 2 * (v - 4)) + (e & 1);
    int n = tn * 16 + (L & 15);
    w_lds[t] = (_Float16)W[(size_t)k * NOUT + n];
  }
  __syncthreads();

  int wid = (blockIdx.x * blockDim.x + threadIdx.x) >> 5;  // global wave = M tile
  int mtiles = (M + 15) >> 4;
  if (wid >= mtiles) return;  // wave-uniform exit, EXEC stays full for WMMA

  int lane = threadIdx.x & 31;
  int l4 = lane & 15, hi = lane >> 4;
  int rowA = wid * 16 + l4;
  if (rowA >= M) rowA = M - 1;  // clamp tail rows
  const float* Arow = A + (size_t)rowA * K;

  v8f c[NT];
#pragma unroll
  for (int tn = 0; tn < NT; ++tn) c[tn] = (v8f){};

#pragma unroll
  for (int kbi = 0; kbi < KT; ++kbi) {
    v16h a;
#pragma unroll
    for (int v = 0; v < 8; ++v) {
      // ISA 16-bit A 16x32 layout: lanes 0-15 K=0..7/16..23, lanes 16-31 K=8..15/24..31
      int k = kbi * 32 + hi * 8 + ((v < 4) ? 2 * v : 16 + 2 * (v - 4));
      float2 av = *(const float2*)(Arow + k);
      a[2 * v]     = (_Float16)av.x;
      a[2 * v + 1] = (_Float16)av.y;
    }
#pragma unroll
    for (int tn = 0; tn < NT; ++tn) {
      v16h b = *(const v16h*)&w_lds[((kbi * NT) + tn) * 512 + lane * 16];
      c[tn] = __builtin_amdgcn_wmma_f32_16x16x32_f16(false, a, false, b, (short)0, c[tn],
                                                     false, false);
    }
  }

  // C/D layout: VGPR r -> M = r (lanes 0-15) / r+8 (lanes 16-31); N = lane&15
  int rbase = wid * 16 + hi * 8;
#pragma unroll
  for (int tn = 0; tn < NT; ++tn) {
    int col = tn * 16 + l4;
#pragma unroll
    for (int r = 0; r < 8; ++r) {
      int row = rbase + r;
      if (row < M) Out[(size_t)row * NOUT + col] = c[tn][r];
    }
  }
}

// ---------------- per-node attention scalars: asrc[i]=h[i].a_s, adst[i]=h[i].a_d ----
template <int D>
__global__ void k_alpha(const float* __restrict__ h, const float* __restrict__ a_s,
                        const float* __restrict__ a_d, float* __restrict__ asrc,
                        float* __restrict__ adst, int n) {
  int wid = (blockIdx.x * blockDim.x + threadIdx.x) >> 5;
  if (wid >= n) return;
  int lane = threadIdx.x & 31;
  const float* row = h + (size_t)wid * D;
  float ps = 0.f, pd = 0.f;
#pragma unroll
  for (int i = lane; i < D; i += 32) {
    float v = row[i];
    ps += v * a_s[i];
    pd += v * a_d[i];
  }
  ps = wave_sum32(ps);
  pd = wave_sum32(pd);
  if (lane == 0) { asrc[wid] = ps; adst[wid] = pd; }
}

// ---------------- per-dst segment softmax + weighted aggregate (wave per node) ------
template <int D, bool RELU>
__global__ void k_aggregate(const float* __restrict__ h, const float* __restrict__ asrc,
                            const float* __restrict__ adst, const int* __restrict__ rp,
                            const int* __restrict__ ci, const float* __restrict__ bias,
                            float* __restrict__ out, int n) {
  constexpr int T = D / 32;  // floats per lane
  int wid = (blockIdx.x * blockDim.x + threadIdx.x) >> 5;
  if (wid >= n) return;
  int lane = threadIdx.x & 31;
  int beg = rp[wid], end = rp[wid + 1];
  float ad = adst[wid];

  // pass 1: segment max (lane-parallel over edges)
  float m = -INFINITY;
  for (int j = beg + lane; j < end; j += 32) {
    float e = asrc[ci[j]] + ad;
    e = (e > 0.f) ? e : 0.2f * e;  // leaky_relu 0.2
    m = fmaxf(m, e);
  }
  m = wave_max32(m);

  // pass 2: exp-sum + weighted feature accumulation (lanes own feature slices)
  float S = 0.f;
  float acc[T];
#pragma unroll
  for (int t = 0; t < T; ++t) acc[t] = 0.f;
  for (int j = beg; j < end; ++j) {
    int s = ci[j];
    float e = asrc[s] + ad;
    e = (e > 0.f) ? e : 0.2f * e;
    float w = __expf(e - m);
    S += w;
    const float* hs = h + (size_t)s * D;
#pragma unroll
    for (int t = 0; t < T; ++t) acc[t] += w * hs[lane + 32 * t];
  }
  float inv = 1.f / (S + 1e-16f);
#pragma unroll
  for (int t = 0; t < T; ++t) {
    float v = acc[t] * inv + bias[lane + 32 * t];
    if (RELU) v = fmaxf(v, 0.f);
    out[(size_t)wid * D + lane + 32 * t] = v;
  }
}

// ---------------- row-wise log_softmax over 32 outputs (in place) -------------------
__global__ void k_logsoftmax32(float* __restrict__ out, int n) {
  int wid = (blockIdx.x * blockDim.x + threadIdx.x) >> 5;
  if (wid >= n) return;
  int lane = threadIdx.x & 31;
  float v = out[(size_t)wid * 32 + lane];
  float m = wave_max32(v);
  float s = wave_sum32(__expf(v - m));
  out[(size_t)wid * 32 + lane] = v - m - logf(s);
}

// ---------------- launcher ----------------
extern "C" void kernel_launch(void* const* d_in, const int* in_sizes, int n_in,
                              void* d_out, int out_size, void* d_ws, size_t ws_size,
                              hipStream_t stream) {
  (void)n_in; (void)out_size; (void)ws_size;
  const float* x      = (const float*)d_in[0];
  const long long* ei = (const long long*)d_in[1];
  const float* W1     = (const float*)d_in[2];
  const float* a_src1 = (const float*)d_in[3];
  const float* a_dst1 = (const float*)d_in[4];
  const float* b1     = (const float*)d_in[5];
  const float* W2     = (const float*)d_in[6];
  const float* a_src2 = (const float*)d_in[7];
  const float* a_dst2 = (const float*)d_in[8];
  const float* b2     = (const float*)d_in[9];
  float* out          = (float*)d_out;

  const int DIN = 128, DHID = 64, DOUT = 32;
  int N = in_sizes[0] / DIN;
  int E = in_sizes[1] / 2;
  const long long* esrc = ei;
  const long long* edst = ei + E;

  // carve workspace
  char* p = (char*)d_ws;
  auto carve = [&](size_t bytes) -> void* {
    void* r = (void*)p;
    p += (bytes + 255) & ~(size_t)255;
    return r;
  };
  float* h1   = (float*)carve((size_t)N * DHID * 4);
  float* out1 = (float*)carve((size_t)N * DHID * 4);
  float* h2   = (float*)carve((size_t)N * DOUT * 4);
  float* asb  = (float*)carve((size_t)N * 4);
  float* adb  = (float*)carve((size_t)N * 4);
  int* deg    = (int*)carve((size_t)N * 4);
  int* rp     = (int*)carve((size_t)(N + 1) * 4);
  int* cur    = (int*)carve((size_t)N * 4);
  int* bsum   = (int*)carve((size_t)4096 * 4);
  int* ci     = (int*)carve((size_t)(E + N) * 4);

  int nb = (N + 1023) / 1024;

  // ---- CSR build (dst-grouped) ----
  k_init_deg<<<(N + 255) / 256, 256, 0, stream>>>(deg, N);
  k_count<<<(E + 255) / 256, 256, 0, stream>>>(edst, E, deg);
  k_scan_block<<<nb, 1024, 0, stream>>>(deg, rp, bsum, N);
  k_scan_bsum<<<1, 32, 0, stream>>>(bsum, nb);
  k_scan_add<<<(N + 255) / 256, 256, 0, stream>>>(rp, bsum, N, E + N);
  k_copy_i32<<<(N + 255) / 256, 256, 0, stream>>>(rp, cur, N);
  k_fill_csr<<<(E + N + 255) / 256, 256, 0, stream>>>(esrc, edst, E, N, cur, ci);

  // ---- layer 1 ----
  {
    int waves = (N + 15) / 16;  // one wave per M tile (owns all N tiles)
    k_gemm_f16wmma<128, 64><<<(waves * 32 + 255) / 256, 256, 0, stream>>>(x, W1, h1, N);
  }
  {
    int blocks = (N * 32 + 255) / 256;
    k_alpha<64><<<blocks, 256, 0, stream>>>(h1, a_src1, a_dst1, asb, adb, N);
    k_aggregate<64, true><<<blocks, 256, 0, stream>>>(h1, asb, adb, rp, ci, b1, out1, N);
  }

  // ---- layer 2 ----
  {
    int waves = (N + 15) / 16;
    k_gemm_f16wmma<64, 32><<<(waves * 32 + 255) / 256, 256, 0, stream>>>(out1, W2, h2, N);
  }
  {
    int blocks = (N * 32 + 255) / 256;
    k_alpha<32><<<blocks, 256, 0, stream>>>(h2, a_src2, a_dst2, asb, adb, N);
    k_aggregate<32, false><<<blocks, 256, 0, stream>>>(h2, asb, adb, rp, ci, b2, out, N);
    k_logsoftmax32<<<blocks, 256, 0, stream>>>(out, N);
  }
}
